// WavGCRN_73770358276448
// MI455X (gfx1250) — matrix operand order repaired and checked by reading, not compiled, not run on hardware
//
#include <hip/hip_runtime.h>
#include <math.h>

typedef __attribute__((ext_vector_type(16))) __bf16 v16bf;
typedef __attribute__((ext_vector_type(8)))  __bf16 v8bf;
typedef __attribute__((ext_vector_type(8)))  float  v8f;

#define NNODE   325
#define BATCH   32
#define HIDDIM  64
#define CCH     68                 // channels in comb = 3 + 1 + 64
#define NBROWS  (NNODE*BATCH)      // 10400
#define GLD     352                // padded row stride for graph matrices
#define GCOLS   (BATCH*CCH)        // 2176
#define TENC    6
#define TDEC    12
#define ALPHA_C 0.05f
#define BETA_C  0.95f
#define GAMMA_C 0.95f
#define INV_SQRT2_C 0.70710678118654752440f

// ---------------- generic bf16 WMMA GEMM ----------------
// C(MxN,f32,row-major,ldc) = scale*(A@B) [+ ax*X] [+ C_old] ; optional bf16 mirror
// Block tile 128x64, 8 wave32 waves in 4x2 grid, 32x32 per wave (4 WMMAs/K-step).
// Double-buffered LDS staging: next tile's global loads overlap current WMMAs,
// one barrier per K-step.
#define BM 128
#define BN 64
#define BK 32
#define LPAD 40   // LDS row length in bf16 (80B, 16B aligned)

__global__ __launch_bounds__(256)
void gemm_bf16_kernel(const __bf16* __restrict__ A, int lda,
                      const __bf16* __restrict__ B, int ldb,
                      float* __restrict__ C, int ldc,
                      __bf16* __restrict__ Cbf,
                      const float* __restrict__ X, float ax,
                      float scale, int accumulate,
                      int M, int N, int K)
{
    __shared__ __bf16 As[2][BM][LPAD];
    __shared__ __bf16 Bt[2][BN][LPAD];   // transposed: Bt[n][k]

    const int tid  = threadIdx.x;
    const int lane = tid & 31;
    const int wid  = tid >> 5;     // 0..7
    const int wm   = wid >> 1;     // 0..3  -> m offset 32*wm
    const int wn   = wid & 1;      // 0..1  -> n offset 32*wn
    const int half = lane >> 4;    // 0/1
    const int l16  = lane & 15;

    const int m0 = blockIdx.y * BM;
    const int n0 = blockIdx.x * BN;

    const int  ntiles = (K + BK - 1) / BK;
    const bool edgeM  = (m0 + BM > M);
    const bool edgeN  = (n0 + BN > N);

    v8f acc[2][2] = {{{}, {}}, {{}, {}}};

    unsigned long long aReg[4];   // 4 chunks of 4 bf16 (A: 128x32 / 256 thr)
    unsigned long long bReg[2];   // 2 chunks of 4 bf16 (B: 32x64 / 256 thr)

    // ---- global -> regs ----
    auto loadA = [&](int kt) {
        const int k0 = kt * BK;
        if (!edgeM && (k0 + BK <= K)) {
#pragma unroll
            for (int i = 0; i < 4; ++i) {
                int c  = tid + i*256;
                int r  = c >> 3;            // 0..127
                int kc = (c & 7) << 2;      // 0..28
                aReg[i] = *(const unsigned long long*)(A + (size_t)(m0 + r) * lda + k0 + kc);
            }
        } else {
#pragma unroll
            for (int i = 0; i < 4; ++i) {
                int c  = tid + i*256;
                int r  = c >> 3;
                int kc = (c & 7) << 2;
                int gm = m0 + r;
                int gk = k0 + kc;
                union { __bf16 h[4]; unsigned long long u; } t;
                t.u = 0ull;
                if (gm < M && gk < K) {
                    const __bf16* p = A + (size_t)gm * lda + gk;
                    if (gk + 4 <= K) {
                        t.u = *(const unsigned long long*)p;
                    } else {
                        for (int j = 0; j < K - gk; ++j) t.h[j] = p[j];
                    }
                }
                aReg[i] = t.u;
            }
        }
    };
    auto loadB = [&](int kt) {
        const int k0 = kt * BK;
        if (!edgeN && (k0 + BK <= K)) {
#pragma unroll
            for (int i = 0; i < 2; ++i) {
                int c  = tid + i*256;
                int kr = c >> 4;            // 0..31
                int nc = (c & 15) << 2;     // 0..60
                bReg[i] = *(const unsigned long long*)(B + (size_t)(k0 + kr) * ldb + n0 + nc);
            }
        } else {
#pragma unroll
            for (int i = 0; i < 2; ++i) {
                int c  = tid + i*256;
                int kr = c >> 4;
                int nc = (c & 15) << 2;
                int gk = k0 + kr;
                int gn = n0 + nc;
                union { __bf16 h[4]; unsigned long long u; } t;
                t.u = 0ull;
                if (gk < K && gn < N) {
                    const __bf16* p = B + (size_t)gk * ldb + gn;
                    if (gn + 4 <= N) {
                        t.u = *(const unsigned long long*)p;
                    } else {
                        for (int j = 0; j < N - gn; ++j) t.h[j] = p[j];
                    }
                }
                bReg[i] = t.u;
            }
        }
    };
    // ---- regs -> LDS ----
    auto storeA = [&](int buf) {
#pragma unroll
        for (int i = 0; i < 4; ++i) {
            int c  = tid + i*256;
            int r  = c >> 3;
            int kc = (c & 7) << 2;
            *(unsigned long long*)&As[buf][r][kc] = aReg[i];
        }
    };
    auto storeB = [&](int buf) {
#pragma unroll
        for (int i = 0; i < 2; ++i) {
            int c  = tid + i*256;
            int kr = c >> 4;
            int nc = (c & 15) << 2;
            union { __bf16 h[4]; unsigned long long u; } t;
            t.u = bReg[i];
            Bt[buf][nc + 0][kr] = t.h[0];
            Bt[buf][nc + 1][kr] = t.h[1];
            Bt[buf][nc + 2][kr] = t.h[2];
            Bt[buf][nc + 3][kr] = t.h[3];
        }
    };

    // ---- fragment loads (ISA 7.12.2 layouts) ----
    // A 16x32: lane m=l16, v[0..7]=A[m][half*8+j], v[8..15]=A[m][16+half*8+j]
    auto load_afrag = [&](int buf, int r) -> v16bf {
        const __bf16* ap = &As[buf][r][0];
        v8bf lo = *(const v8bf*)(ap + half*8);
        v8bf hi = *(const v8bf*)(ap + 16 + half*8);
        return __builtin_shufflevector(lo, hi,
            0,1,2,3,4,5,6,7,8,9,10,11,12,13,14,15);
    };
    // B 32x16: lane n=l16, v[0..7]=B[half*8+j][n], v[8..15]=B[16+half*8+j][n] (Bt[n][*])
    auto load_bfrag = [&](int buf, int n) -> v16bf {
        const __bf16* bp = &Bt[buf][n][0];
        v8bf lo = *(const v8bf*)(bp + half*8);
        v8bf hi = *(const v8bf*)(bp + 16 + half*8);
        return __builtin_shufflevector(lo, hi,
            0,1,2,3,4,5,6,7,8,9,10,11,12,13,14,15);
    };

    // prologue: stage tile 0
    loadA(0); loadB(0);
    storeA(0); storeB(0);
    __syncthreads();

    for (int kt = 0; kt < ntiles; ++kt) {
        const int buf = kt & 1;
        const bool more = (kt + 1 < ntiles);

        // issue next tile's global loads; LOADcnt wait deferred to LDS stores
        if (more) { loadA(kt + 1); loadB(kt + 1); }

        v16bf a0 = load_afrag(buf, wm*32 + l16);
        v16bf a1 = load_afrag(buf, wm*32 + 16 + l16);
        v16bf b0 = load_bfrag(buf, wn*32 + l16);
        v16bf b1 = load_bfrag(buf, wn*32 + 16 + l16);

        acc[0][0] = __builtin_amdgcn_wmma_f32_16x16x32_bf16(
            false, a0, false, b0, (short)0, acc[0][0], false, false);
        acc[0][1] = __builtin_amdgcn_wmma_f32_16x16x32_bf16(
            false, a0, false, b1, (short)0, acc[0][1], false, false);
        acc[1][0] = __builtin_amdgcn_wmma_f32_16x16x32_bf16(
            false, a1, false, b0, (short)0, acc[1][0], false, false);
        acc[1][1] = __builtin_amdgcn_wmma_f32_16x16x32_bf16(
            false, a1, false, b1, (short)0, acc[1][1], false, false);

        if (more) {
            storeA(buf ^ 1);
            storeB(buf ^ 1);
            __syncthreads();
        }
    }

    // ---- epilogue: C/D layout: VGPR j -> row (half*8 + j), col l16 ----
#pragma unroll
    for (int ti = 0; ti < 2; ++ti) {
        const int mbase = m0 + wm*32 + ti*16 + half*8;
#pragma unroll
        for (int tj = 0; tj < 2; ++tj) {
            const int ncol = n0 + wn*32 + tj*16 + l16;
            if (ncol >= N) continue;
            v8f a = acc[ti][tj];
#pragma unroll
            for (int j = 0; j < 8; ++j) {
                int row = mbase + j;
                if (row >= M) continue;
                size_t idx = (size_t)row * ldc + ncol;
                float o = scale * a[j];
                if (X)          o += ax * X[idx];
                if (accumulate) o += C[idx];
                C[idx] = o;
                if (Cbf) Cbf[idx] = (__bf16)o;
            }
        }
    }
}

// ---------------- prep / elementwise kernels ----------------
__global__ void rowsum_kernel(const float* __restrict__ adj, float* rs1, float* rsT)
{
    __shared__ float s1[256];
    __shared__ float s2[256];
    int v = blockIdx.x;
    int tid = threadIdx.x;
    float a = 0.f, b = 0.f;
    for (int w2 = tid; w2 < NNODE; w2 += 256) {
        a += adj[(size_t)v * NNODE + w2];
        b += adj[(size_t)w2 * NNODE + v];
    }
    s1[tid] = a; s2[tid] = b;
    __syncthreads();
    for (int off = 128; off > 0; off >>= 1) {
        if (tid < off) { s1[tid] += s1[tid + off]; s2[tid] += s2[tid + off]; }
        __syncthreads();
    }
    if (tid == 0) { rs1[v] = s1[0] + 1.f; rsT[v] = s2[0] + 1.f; }
}

// M_A[w][v] = beta*A1[v][w] + gamma*P[v][w];  M_T[w][v] = beta*A1T[v][w] + gamma*P[v][w]
__global__ void build_M_kernel(const float* __restrict__ adj, const float* __restrict__ P,
                               const float* __restrict__ rs1, const float* __restrict__ rsT,
                               __bf16* MA, __bf16* MT)
{
    int i = blockIdx.x * 256 + threadIdx.x;
    if (i >= GLD * GLD) return;
    int wn = i / GLD;
    int v  = i % GLD;
    float ma = 0.f, mt = 0.f;
    if (v < NNODE && wn < NNODE) {
        float d  = (v == wn) ? 1.f : 0.f;
        float pa = P[(size_t)v * NNODE + wn];
        ma = BETA_C * ((adj[(size_t)v  * NNODE + wn] + d) / rs1[v]) + GAMMA_C * pa;
        mt = BETA_C * ((adj[(size_t)wn * NNODE + v ] + d) / rsT[v]) + GAMMA_C * pa;
    }
    MA[(size_t)wn * GLD + v] = (__bf16)ma;
    MT[(size_t)wn * GLD + v] = (__bf16)mt;
}

__global__ void cvt_bf16_kernel(const float* __restrict__ in, __bf16* __restrict__ out, int n)
{
    int i = blockIdx.x * 256 + threadIdx.x;
    if (i < n) out[i] = (__bf16)in[i];
}

__global__ void time_feat_kernel(const float* __restrict__ st, float* tds, float* tod)
{
    int i = blockIdx.x * 256 + threadIdx.x;
    if (i >= BATCH * TDEC) return;
    int t = i % TDEC, b = i / TDEC;
    float hour   = (st[(b*4 + 2)*12 + t] + 0.5f) * 23.f;
    float minute = (st[(b*4 + 3)*12 + t] + 0.5f) * 59.f;
    float hidx = floorf((hour * 60.f + minute) / 5.f);   // 60/PPH = 5
    if ((t & 1) == 0) tds[b*TENC + (t >> 1)] = hidx;
    tod[b*TDEC + t] = floorf((st[(b*4 + 2)*12 + t] + 0.5f) * 6.f);
}

// (B,N,H) -> node-major [v][b][h]
__global__ void transpose_state_kernel(const float* __restrict__ in, float* __restrict__ out, int n)
{
    int i = blockIdx.x * 256 + threadIdx.x;
    if (i >= n) return;
    int h = i % HIDDIM;
    int t = i / HIDDIM;
    int v = t % NNODE;
    int b = t / NNODE;
    out[((size_t)(v*BATCH + b))*HIDDIM + h] = in[i];
}

__global__ void build_comb_enc_kernel(const float* __restrict__ x, const float* __restrict__ tds,
                                      const float* __restrict__ H, float sign, int tstep,
                                      float* __restrict__ combF, __bf16* __restrict__ combB)
{
    int i = blockIdx.x * 256 + threadIdx.x;
    if (i >= NBROWS * CCH) return;
    int c  = i % CCH;
    int vb = i / CCH;
    int b = vb % BATCH, v = vb / BATCH;
    float val;
    if (c < 3) {
        const float* xp = x + (((size_t)b*3 + c)*NNODE + v)*12 + 2*tstep;
        val = (xp[0] + sign * xp[1]) * INV_SQRT2_C;
    } else if (c == 3) {
        val = tds[b*TENC + tstep];
    } else {
        val = H[(size_t)vb*HIDDIM + (c - 4)];
    }
    combF[i] = val;
    combB[i] = (__bf16)val;
}

__global__ void build_comb_dec_kernel(const float* __restrict__ din, const float* __restrict__ tod,
                                      const float* __restrict__ H, int tstep,
                                      float* __restrict__ combF, __bf16* __restrict__ combB)
{
    int i = blockIdx.x * 256 + threadIdx.x;
    if (i >= NBROWS * CCH) return;
    int c  = i % CCH;
    int vb = i / CCH;
    int b = vb % BATCH;
    float val;
    if (c < 3)       val = din[(size_t)vb*3 + c];
    else if (c == 3) val = tod[b*TDEC + tstep];
    else             val = H[(size_t)vb*HIDDIM + (c - 4)];
    combF[i] = val;
    combB[i] = (__bf16)val;
}

__global__ void sigmoid_bias2_kernel(float* __restrict__ buf, const float* __restrict__ ba,
                                     const float* __restrict__ bb, int n)
{
    int i = blockIdx.x * 256 + threadIdx.x;
    if (i >= n) return;
    int c = i & 63;
    float v = buf[i] + ba[c] + bb[c];
    buf[i] = 1.f / (1.f + __expf(-v));
}

__global__ void build_temp_kernel(float* __restrict__ combF, __bf16* __restrict__ combB,
                                  const float* __restrict__ r, const float* __restrict__ H)
{
    int i = blockIdx.x * 256 + threadIdx.x;
    if (i >= NBROWS * CCH) return;
    int c = i % CCH;
    if (c < 4) return;                        // xin channels unchanged
    int vb = i / CCH;
    float val = r[(size_t)vb*HIDDIM + (c-4)] * H[(size_t)vb*HIDDIM + (c-4)];
    combF[i] = val;
    combB[i] = (__bf16)val;
}

__global__ void update_kernel(const float* __restrict__ cpre, const float* __restrict__ ba,
                              const float* __restrict__ bb, const float* __restrict__ z,
                              float* __restrict__ H, float* __restrict__ C, int n)
{
    int i = blockIdx.x * 256 + threadIdx.x;
    if (i >= n) return;
    int c = i & 63;
    float cand = tanhf(cpre[i] + ba[c] + bb[c]);
    float zi = z[i];
    float h  = zi * H[i] + (1.f - zi) * cand;
    H[i] = h;
    C[i] = cand;
}

__global__ void fuse_kernel(const float* __restrict__ S1, const float* __restrict__ S2,
                            const float* __restrict__ wL, const float* __restrict__ bL,
                            const float* __restrict__ wH, const float* __restrict__ bH,
                            __bf16* __restrict__ Hm, __bf16* __restrict__ cat)
{
    int i = blockIdx.x * 256 + threadIdx.x;
    if (i >= NBROWS * HIDDIM) return;
    int h  = i % HIDDIM;
    int vb = i / HIDDIM;
    int v  = vb / BATCH;
    float a  = S1[i], b2 = S2[i];
    float s  = a + b2;
    float L  = s * wL[(size_t)v*HIDDIM + h] + 2.f * bL[(size_t)v*HIDDIM + h];
    float Hh = s * wH[(size_t)v*HIDDIM + h] + 2.f * bH[(size_t)v*HIDDIM + h];
    Hm[(size_t)vb*128 + 2*h]     = (__bf16)L;
    Hm[(size_t)vb*128 + 2*h + 1] = (__bf16)Hh;
    cat[(size_t)vb*128 + h]      = (__bf16)a;
    cat[(size_t)vb*128 + 64 + h] = (__bf16)b2;
}

__global__ void add_bias2_kernel(float* __restrict__ buf, const float* __restrict__ ba,
                                 const float* __restrict__ bb, float sa, float sb, int n)
{
    int i = blockIdx.x * 256 + threadIdx.x;
    if (i >= n) return;
    int c = i & 63;
    buf[i] += sa * ba[c] + sb * bb[c];
}

__global__ void zero_kernel(float* p, int n)
{
    int i = blockIdx.x * 256 + threadIdx.x;
    if (i < n) p[i] = 0.f;
}

// out = H @ W_fc + b_fc ; also scatters into final output tensor and din
__global__ void fc_kernel(const float* __restrict__ H, const float* __restrict__ Wfc,
                          const float* __restrict__ bfc, float* __restrict__ din,
                          float* __restrict__ out, int tstep)
{
    int vb = blockIdx.x * 256 + threadIdx.x;
    if (vb >= NBROWS) return;
    int b = vb % BATCH, v = vb / BATCH;
    const float* hr = H + (size_t)vb * HIDDIM;
#pragma unroll
    for (int k = 0; k < 3; ++k) {
        float s = bfc[k];
        for (int h = 0; h < HIDDIM; ++h) s += hr[h] * Wfc[h*3 + k];
        din[(size_t)vb*3 + k] = s;
        int e = tstep*3 + k;
        int c = e / 12, l = e % 12;
        out[(((size_t)b*3 + c)*NNODE + v)*12 + l] = s;
    }
}

// ---------------- host orchestration ----------------
static inline void launch_gemm(hipStream_t s, const __bf16* A, int lda, const __bf16* B, int ldb,
                               float* C, int ldc, __bf16* Cbf, const float* X, float ax,
                               float scale, int acc, int M, int N, int K)
{
    dim3 grid((N + BN - 1) / BN, (M + BM - 1) / BM);
    gemm_bf16_kernel<<<grid, 256, 0, s>>>(A, lda, B, ldb, C, ldc, Cbf, X, ax, scale, acc, M, N, K);
}

struct Ws {
    __bf16 *MA, *MT;
    __bf16 *WencD, *WencAD, *Wdec, *WaggH, *WaggC, *WcamH, *WcamC;
    float  *tds, *tod, *rs1, *rsT;
    float  *H1, *C1, *H2, *C2, *Hd, *Cd;
    float  *combF; __bf16 *combB;
    float  *hF[4]; __bf16 *hB[4];
    float  *zb, *rb, *cb;
    __bf16 *HmB, *catB;
    float  *din;
};

static void proj6(hipStream_t st, Ws& w, const __bf16* Wbf, int gA, int gT, float* out)
{
    const __bf16* WA = Wbf + (size_t)gA * 204 * 64;
    const __bf16* WT = Wbf + (size_t)gT * 204 * 64;
    const __bf16* sA[3] = { w.combB, w.hB[0], w.hB[1] };
    const __bf16* sT[3] = { w.combB, w.hB[2], w.hB[3] };
    for (int p = 0; p < 3; ++p)
        launch_gemm(st, sA[p], CCH, WA + (size_t)p*CCH*64, 64, out, 64,
                    nullptr, nullptr, 0.f, 1.f, p ? 1 : 0, NBROWS, 64, CCH);
    for (int p = 0; p < 3; ++p)
        launch_gemm(st, sT[p], CCH, WT + (size_t)p*CCH*64, 64, out, 64,
                    nullptr, nullptr, 0.f, 1.f, 1, NBROWS, 64, CCH);
}

static void diffuse4(hipStream_t st, Ws& w)
{
    // hA1 = a*comb + MA@comb ; hA2 = a*comb + MA@hA1 ; hT1/hT2 with MT
    launch_gemm(st, w.MA, GLD, w.combB, GCOLS, w.hF[0], GCOLS, w.hB[0], w.combF, ALPHA_C, 1.f, 0, NNODE, GCOLS, NNODE);
    launch_gemm(st, w.MA, GLD, w.hB[0], GCOLS, w.hF[1], GCOLS, w.hB[1], w.combF, ALPHA_C, 1.f, 0, NNODE, GCOLS, NNODE);
    launch_gemm(st, w.MT, GLD, w.combB, GCOLS, w.hF[2], GCOLS, w.hB[2], w.combF, ALPHA_C, 1.f, 0, NNODE, GCOLS, NNODE);
    launch_gemm(st, w.MT, GLD, w.hB[2], GCOLS, w.hF[3], GCOLS, w.hB[3], w.combF, ALPHA_C, 1.f, 0, NNODE, GCOLS, NNODE);
}

static void cell_step(hipStream_t st, Ws& w, const __bf16* Wbf, const float* bdev,
                      float* H, float* C)
{
    const int nH = NBROWS * HIDDIM;
    const int gH = (nH + 255) / 256;
    const int nC = NBROWS * CCH;
    const int gC = (nC + 255) / 256;

    diffuse4(st, w);
    proj6(st, w, Wbf, 0, 1, w.zb);
    proj6(st, w, Wbf, 2, 3, w.rb);
    sigmoid_bias2_kernel<<<gH, 256, 0, st>>>(w.zb, bdev + 0*64, bdev + 1*64, nH);
    sigmoid_bias2_kernel<<<gH, 256, 0, st>>>(w.rb, bdev + 2*64, bdev + 3*64, nH);
    build_temp_kernel<<<gC, 256, 0, st>>>(w.combF, w.combB, w.rb, H);
    diffuse4(st, w);
    proj6(st, w, Wbf, 4, 5, w.cb);
    update_kernel<<<gH, 256, 0, st>>>(w.cb, bdev + 4*64, bdev + 5*64, w.zb, H, C, nH);
}

extern "C" void kernel_launch(void* const* d_in, const int* in_sizes, int n_in,
                              void* d_out, int out_size, void* d_ws, size_t ws_size,
                              hipStream_t stream)
{
    (void)in_sizes; (void)n_in; (void)out_size; (void)ws_size;
    const float* x      = (const float*)d_in[0];
    const float* st_    = (const float*)d_in[1];
    const float* adj    = (const float*)d_in[2];
    const float* P      = (const float*)d_in[3];
    const float* H1_0   = (const float*)d_in[4];
    const float* C1_0   = (const float*)d_in[5];
    const float* H2_0   = (const float*)d_in[6];
    const float* C2_0   = (const float*)d_in[7];
    const float* W_encD = (const float*)d_in[8];
    const float* b_encD = (const float*)d_in[9];
    const float* W_encAD= (const float*)d_in[10];
    const float* b_encAD= (const float*)d_in[11];
    const float* W_dec  = (const float*)d_in[12];
    const float* b_dec  = (const float*)d_in[13];
    const float* W_fc   = (const float*)d_in[14];
    const float* b_fc   = (const float*)d_in[15];
    const float* W_aggH = (const float*)d_in[16];
    const float* b_aggH = (const float*)d_in[17];
    const float* W_aggC = (const float*)d_in[18];
    const float* b_aggC = (const float*)d_in[19];
    const float* W_camH = (const float*)d_in[20];
    const float* b_camH = (const float*)d_in[21];
    const float* W_camC = (const float*)d_in[22];
    const float* b_camC = (const float*)d_in[23];
    const float* wL     = (const float*)d_in[24];
    const float* bL     = (const float*)d_in[25];
    const float* wH     = (const float*)d_in[26];
    const float* bH     = (const float*)d_in[27];
    float* out = (float*)d_out;

    // deterministic bump allocator over workspace
    char* wp = (char*)d_ws;
    auto alloc = [&](size_t bytes) -> void* {
        void* p = (void*)wp;
        wp += (bytes + 255) & ~(size_t)255;
        return p;
    };

    Ws w;
    w.MA     = (__bf16*)alloc((size_t)GLD*GLD*2);
    w.MT     = (__bf16*)alloc((size_t)GLD*GLD*2);
    w.WencD  = (__bf16*)alloc((size_t)6*204*64*2);
    w.WencAD = (__bf16*)alloc((size_t)6*204*64*2);
    w.Wdec   = (__bf16*)alloc((size_t)6*204*64*2);
    w.WaggH  = (__bf16*)alloc((size_t)128*64*2);
    w.WaggC  = (__bf16*)alloc((size_t)128*64*2);
    w.WcamH  = (__bf16*)alloc((size_t)128*64*2);
    w.WcamC  = (__bf16*)alloc((size_t)128*64*2);
    w.tds    = (float*)alloc((size_t)BATCH*TENC*4);
    w.tod    = (float*)alloc((size_t)BATCH*TDEC*4);
    w.rs1    = (float*)alloc((size_t)NNODE*4);
    w.rsT    = (float*)alloc((size_t)NNODE*4);
    w.H1     = (float*)alloc((size_t)NBROWS*HIDDIM*4);
    w.C1     = (float*)alloc((size_t)NBROWS*HIDDIM*4);
    w.H2     = (float*)alloc((size_t)NBROWS*HIDDIM*4);
    w.C2     = (float*)alloc((size_t)NBROWS*HIDDIM*4);
    w.Hd     = (float*)alloc((size_t)NBROWS*HIDDIM*4);
    w.Cd     = (float*)alloc((size_t)NBROWS*HIDDIM*4);
    w.combF  = (float*)alloc((size_t)NBROWS*CCH*4);
    w.combB  = (__bf16*)alloc((size_t)NBROWS*CCH*2);
    for (int i = 0; i < 4; ++i) {
        w.hF[i] = (float*)alloc((size_t)NBROWS*CCH*4);
        w.hB[i] = (__bf16*)alloc((size_t)NBROWS*CCH*2);
    }
    w.zb   = (float*)alloc((size_t)NBROWS*HIDDIM*4);
    w.rb   = (float*)alloc((size_t)NBROWS*HIDDIM*4);
    w.cb   = (float*)alloc((size_t)NBROWS*HIDDIM*4);
    w.HmB  = (__bf16*)alloc((size_t)NBROWS*128*2);
    w.catB = (__bf16*)alloc((size_t)NBROWS*128*2);
    w.din  = (float*)alloc((size_t)NBROWS*3*4);

    // ---- prep ----
    rowsum_kernel<<<NNODE, 256, 0, stream>>>(adj, w.rs1, w.rsT);
    build_M_kernel<<<(GLD*GLD + 255)/256, 256, 0, stream>>>(adj, P, w.rs1, w.rsT, w.MA, w.MT);
    {
        int nw = 6*204*64, gw = (nw + 255)/256;
        cvt_bf16_kernel<<<gw, 256, 0, stream>>>(W_encD,  w.WencD,  nw);
        cvt_bf16_kernel<<<gw, 256, 0, stream>>>(W_encAD, w.WencAD, nw);
        cvt_bf16_kernel<<<gw, 256, 0, stream>>>(W_dec,   w.Wdec,   nw);
        int nm = 128*64, gm = (nm + 255)/256;
        cvt_bf16_kernel<<<gm, 256, 0, stream>>>(W_aggH, w.WaggH, nm);
        cvt_bf16_kernel<<<gm, 256, 0, stream>>>(W_aggC, w.WaggC, nm);
        cvt_bf16_kernel<<<gm, 256, 0, stream>>>(W_camH, w.WcamH, nm);
        cvt_bf16_kernel<<<gm, 256, 0, stream>>>(W_camC, w.WcamC, nm);
    }
    time_feat_kernel<<<(BATCH*TDEC + 255)/256, 256, 0, stream>>>(st_, w.tds, w.tod);
    {
        int ns = BATCH*NNODE*HIDDIM, gs = (ns + 255)/256;
        transpose_state_kernel<<<gs, 256, 0, stream>>>(H1_0, w.H1, ns);
        transpose_state_kernel<<<gs, 256, 0, stream>>>(C1_0, w.C1, ns);
        transpose_state_kernel<<<gs, 256, 0, stream>>>(H2_0, w.H2, ns);
        transpose_state_kernel<<<gs, 256, 0, stream>>>(C2_0, w.C2, ns);
    }

    const int nC = NBROWS * CCH;
    const int gC = (nC + 255)/256;
    const int nH = NBROWS * HIDDIM;
    const int gH = (nH + 255)/256;

    // ---- encoder D (detail: xe - xo) ----
    for (int t = 0; t < TENC; ++t) {
        build_comb_enc_kernel<<<gC, 256, 0, stream>>>(x, w.tds, w.H1, -1.f, t, w.combF, w.combB);
        cell_step(stream, w, w.WencD, b_encD, w.H1, w.C1);
    }
    // ---- encoder AD (approx: xe + xo) ----
    for (int t = 0; t < TENC; ++t) {
        build_comb_enc_kernel<<<gC, 256, 0, stream>>>(x, w.tds, w.H2, +1.f, t, w.combF, w.combB);
        cell_step(stream, w, w.WencAD, b_encAD, w.H2, w.C2);
    }

    // ---- fusion / aggregation ----
    fuse_kernel<<<gH, 256, 0, stream>>>(w.H1, w.H2, wL, bL, wH, bH, w.HmB, w.catB);
    launch_gemm(stream, w.HmB, 128, w.WcamH, 64, w.Hd, 64, nullptr, nullptr, 0.f, 0.3f, 0, NBROWS, 64, 128);
    launch_gemm(stream, w.catB, 128, w.WaggH, 64, w.Hd, 64, nullptr, nullptr, 0.f, 0.7f, 1, NBROWS, 64, 128);
    add_bias2_kernel<<<gH, 256, 0, stream>>>(w.Hd, b_camH, b_aggH, 0.3f, 0.7f, nH);

    fuse_kernel<<<gH, 256, 0, stream>>>(w.C1, w.C2, wL, bL, wH, bH, w.HmB, w.catB);
    launch_gemm(stream, w.HmB, 128, w.WcamC, 64, w.Cd, 64, nullptr, nullptr, 0.f, 0.3f, 0, NBROWS, 64, 128);
    launch_gemm(stream, w.catB, 128, w.WaggC, 64, w.Cd, 64, nullptr, nullptr, 0.f, 0.7f, 1, NBROWS, 64, 128);
    add_bias2_kernel<<<gH, 256, 0, stream>>>(w.Cd, b_camC, b_aggC, 0.3f, 0.7f, nH);

    // ---- decoder ----
    zero_kernel<<<(NBROWS*3 + 255)/256, 256, 0, stream>>>(w.din, NBROWS*3);
    for (int t = 0; t < TDEC; ++t) {
        build_comb_dec_kernel<<<gC, 256, 0, stream>>>(w.din, w.tod, w.Hd, t, w.combF, w.combB);
        cell_step(stream, w, w.Wdec, b_dec, w.Hd, w.Cd);
        fc_kernel<<<(NBROWS + 255)/256, 256, 0, stream>>>(w.Hd, W_fc, b_fc, w.din, out, t);
    }
}
